// VARS_S_Block_18098992185941
// MI455X (gfx1250) — compile-verified, matrix operand order checked by reading, not run on hardware
//
#include <hip/hip_runtime.h>
#include <math.h>

// ---------------- types ----------------
typedef __attribute__((ext_vector_type(8)))  __bf16 v8bf;
typedef __attribute__((ext_vector_type(16))) __bf16 v16bf;
typedef __attribute__((ext_vector_type(8)))  float  v8f;

// ---------------- constants ----------------
constexpr int kH   = 14;
constexpr int kHF  = 8;     // H/2+1
constexpr int kNP  = 196;   // H*H
constexpr int kDIM = 768;
constexpr int kINT = 5;
constexpr int kB   = 32;
constexpr int kM   = kB * kNP;   // 6272 rows
constexpr int kHID = 4 * kDIM;   // 3072
constexpr float kLAM = 3.5f;

// GEMM tiling
#define GBM 128
#define GBN 128
#define GBK 64
#define GKP 72   // padded K stride in LDS halves (144B rows: 16B aligned, conflict-free)

// ---------------- helpers ----------------
__device__ inline v16bf load_frag(const __bf16* base, int lane, int slab) {
  // 16-bit A/B fragment layout (wave32): lane group g = lane>>4, row/col = lane&15
  // elements 0..7  -> k = slab + g*8 .. +7        (contiguous 16B)
  // elements 8..15 -> k = slab + 16 + g*8 .. +7   (contiguous 16B)
  int grp = lane >> 4;
  int r   = lane & 15;
  const __bf16* p = base + r * GKP + slab + grp * 8;
  v8bf lo = *(const v8bf*)(p);
  v8bf hi = *(const v8bf*)(p + 16);
  v16bf out;
#pragma unroll
  for (int i = 0; i < 8; ++i) { out[i] = lo[i]; out[i + 8] = hi[i]; }
  return out;
}

__device__ inline float softf(float x, float t) {
  float a = fabsf(x) - t;
  return a > 0.0f ? copysignf(a, x) : 0.0f;
}

// Async DMA of one 128x64 bf16 tile (global, row stride ldK) into LDS tile
// (row stride GKP halves). 1024 x 16B chunks, 4 per thread -> 4 async ops/thread.
__device__ inline void async_tile_load(const __bf16* gbase, int ldK, int rowBase,
                                       int k0, __bf16* lbase) {
  int tid = threadIdx.x;
#pragma unroll
  for (int i = 0; i < 4; ++i) {
    int t  = tid + i * 256;
    int r  = t >> 3;            // 0..127
    int kc = (t & 7) * 8;       // 0..56
    const __bf16* g = gbase + (size_t)(rowBase + r) * ldK + k0 + kc;
    unsigned l = (unsigned)(unsigned long long)(const void*)(lbase + r * GKP + kc);
    asm volatile("global_load_async_to_lds_b128 %0, %1, off"
                 :: "v"(l), "v"(g) : "memory");
  }
}

// ---------------- fp32 -> bf16 convert ----------------
__global__ void cvt_bf16(const float* __restrict__ in, __bf16* __restrict__ out, int n) {
  int i = blockIdx.x * blockDim.x + threadIdx.x;
  if (i < n) out[i] = (__bf16)in[i];
}

// ---------------- WMMA GEMM: out = A(MxK) @ W(NxK)^T  (+epilogue) ----------------
// mode 0: outF fp32, transposed write to (B, C, Np) layout (for conv input)
// mode 1: outF fp32, out = resid + acc + bias          (row-major M x N)
// mode 2: outH bf16, out = gelu(acc + bias)            (row-major M x N)
__global__ __launch_bounds__(256) void gemm_wmma(
    const __bf16* __restrict__ A, const __bf16* __restrict__ W,
    const float* __restrict__ bias, const float* __restrict__ resid,
    float* __restrict__ outF, __bf16* __restrict__ outH,
    int M, int N, int K, int mode)
{
  __shared__ __align__(16) __bf16 As[2][GBM][GKP];
  __shared__ __align__(16) __bf16 Bs[2][GBN][GKP];

  const int tid  = threadIdx.x;
  const int lane = tid & 31;
  const int wv   = tid >> 5;       // 8 waves
  const int wm   = wv & 3;         // 4 waves along M (32 rows each)
  const int wn   = wv >> 2;        // 2 waves along N (64 cols each)
  const int rowBase = blockIdx.y * GBM;
  const int colBase = blockIdx.x * GBN;

  v8f acc[8];
#pragma unroll
  for (int i = 0; i < 8; ++i) acc[i] = (v8f)(0.0f);

  // preload tile 0 (8 async ops per thread outstanding)
  async_tile_load(A, K, rowBase, 0, &As[0][0][0]);
  async_tile_load(W, K, colBase, 0, &Bs[0][0][0]);

  const int nTiles = K / GBK;
  for (int t = 0; t < nTiles; ++t) {
    const int cur = t & 1;
    const int nxt = cur ^ 1;
    if (t + 1 < nTiles) {
      // stream next tile while computing current one
      async_tile_load(A, K, rowBase, (t + 1) * GBK, &As[nxt][0][0]);
      async_tile_load(W, K, colBase, (t + 1) * GBK, &Bs[nxt][0][0]);
      // async loads complete in order: <=8 outstanding => current tile landed
      asm volatile("s_wait_asynccnt 0x8" ::: "memory");
    } else {
      asm volatile("s_wait_asynccnt 0x0" ::: "memory");
    }
    __syncthreads();

    const __bf16* aBase = &As[cur][wm * 32][0];
    const __bf16* bBase = &Bs[cur][wn * 64][0];
#pragma unroll
    for (int slab = 0; slab < GBK; slab += 32) {
      v16bf a0 = load_frag(aBase, lane, slab);
      v16bf a1 = load_frag(aBase + 16 * GKP, lane, slab);
#pragma unroll
      for (int fr = 0; fr < 4; ++fr) {
        v16bf bfr = load_frag(bBase + fr * 16 * GKP, lane, slab);
        acc[fr] = __builtin_amdgcn_wmma_f32_16x16x32_bf16(
            false, a0, false, bfr, (short)0, acc[fr], false, false);
        acc[4 + fr] = __builtin_amdgcn_wmma_f32_16x16x32_bf16(
            false, a1, false, bfr, (short)0, acc[4 + fr], false, false);
      }
    }
    __syncthreads();
  }

  // epilogue: C/D layout: VGPR j -> row j + 8*(lane>>4); col = lane&15
  const int grp = lane >> 4;
  const int cc  = lane & 15;
#pragma unroll
  for (int ar = 0; ar < 2; ++ar) {
#pragma unroll
    for (int fr = 0; fr < 4; ++fr) {
      int col = colBase + wn * 64 + fr * 16 + cc;
#pragma unroll
      for (int j = 0; j < 8; ++j) {
        int m = rowBase + wm * 32 + ar * 16 + grp * 8 + j;
        float v = acc[ar * 4 + fr][j];
        if (mode == 0) {
          int b = m / kNP, ns = m % kNP;
          outF[((size_t)b * kDIM + col) * kNP + ns] = v;
        } else if (mode == 1) {
          v += bias[col] + resid[(size_t)m * N + col];
          outF[(size_t)m * N + col] = v;
        } else {
          v += bias[col];
          v = 0.5f * v * (1.0f + erff(v * 0.70710678118654752f));
          outH[(size_t)m * N + col] = (__bf16)v;
        }
      }
    }
  }
}

// ---------------- twiddle table (14-point) ----------------
__device__ inline void build_tw(float* tc, float* tsn) {
  if (threadIdx.x < kH) {
    float th = (6.283185307179586f / 14.0f) * (float)threadIdx.x;
    tc[threadIdx.x]  = cosf(th);
    tsn[threadIdx.x] = sinf(th);
  }
}

// ---------------- kk = k_i * conj(k_j) per group/bin ----------------
__global__ __launch_bounds__(256) void kk_compute(const float* __restrict__ kern,
                                                  float* __restrict__ kk)
{
  int g = blockIdx.x;
  for (int e = threadIdx.x; e < kINT * kINT * 112; e += 256) {
    int i = e / (kINT * 112);
    int rem = e % (kINT * 112);
    int j = rem / 112;
    int hf = rem % 112;
    size_t ia = ((size_t)g * kINT + i) * 224 + (size_t)hf * 2;
    size_t ja = ((size_t)g * kINT + j) * 224 + (size_t)hf * 2;
    float ar = kern[ia], ai = kern[ia + 1];
    float br = kern[ja], bi = kern[ja + 1];
    size_t o = ((size_t)g * 25 + i * 5 + j) * 224 + (size_t)hf * 2;
    kk[o]     = ar * br + ai * bi;   // Re(a * conj(b))
    kk[o + 1] = ai * br - ar * bi;   // Im(a * conj(b))
  }
}

// ---------------- forward conv: u = circconv(vx, k); s = soft(u, LAM) ----------------
__global__ __launch_bounds__(128) void conv_fwd(const float* __restrict__ vx,
                                                const float* __restrict__ kern,
                                                float* __restrict__ u,
                                                float* __restrict__ s)
{
  int b = blockIdx.x / kDIM, g = blockIdx.x % kDIM;
  __shared__ float tc[kH], tsn[kH];
  __shared__ float xs[kH][kH];
  __shared__ float Tr[kH][kHF], Tq[kH][kHF];
  __shared__ float Xr[kH][kHF], Xq[kH][kHF];
  __shared__ float Yr[kINT][kH][kHF], Yq[kINT][kH][kHF];
  __shared__ float Ar[kINT][kH][kHF], Aq[kINT][kH][kHF];
  int tid = threadIdx.x;
  build_tw(tc, tsn);
  const float* xg = vx + ((size_t)b * kDIM + g) * kNP;
  for (int e = tid; e < kNP; e += 128) xs[e / kH][e % kH] = xg[e];
  __syncthreads();
  // rfft along columns
  for (int e = tid; e < 112; e += 128) {
    int r = e >> 3, f = e & 7;
    float sr = 0, si = 0;
    for (int c = 0; c < kH; ++c) {
      int m = (f * c) % kH;
      float xv = xs[r][c];
      sr += xv * tc[m]; si -= xv * tsn[m];
    }
    Tr[r][f] = sr; Tq[r][f] = si;
  }
  __syncthreads();
  // fft along rows, ortho scale 1/14
  for (int e = tid; e < 112; e += 128) {
    int h = e >> 3, f = e & 7;
    float sr = 0, si = 0;
    for (int r = 0; r < kH; ++r) {
      int m = (h * r) % kH;
      float a = Tr[r][f], c2 = Tq[r][f];
      sr += a * tc[m] + c2 * tsn[m];
      si += c2 * tc[m] - a * tsn[m];
    }
    Xr[h][f] = sr * (1.0f / 14.0f);
    Xq[h][f] = si * (1.0f / 14.0f);
  }
  __syncthreads();
  // spectrum multiply with k[g][i]
  for (int e = tid; e < kINT * 112; e += 128) {
    int i = e / 112, hf = e % 112;
    int h = hf >> 3, f = hf & 7;
    size_t ki = ((size_t)g * kINT + i) * 224 + (size_t)hf * 2;
    float kr = kern[ki], kq = kern[ki + 1];
    float xr = Xr[h][f], xq = Xq[h][f];
    Yr[i][h][f] = kr * xr - kq * xq;
    Yq[i][h][f] = kr * xq + kq * xr;
  }
  __syncthreads();
  // ifft along rows
  for (int e = tid; e < kINT * 112; e += 128) {
    int i = e / 112, rf = e % 112;
    int r = rf >> 3, f = rf & 7;
    float sr = 0, si = 0;
    for (int h = 0; h < kH; ++h) {
      int m = (h * r) % kH;
      float a = Yr[i][h][f], c2 = Yq[i][h][f];
      sr += a * tc[m] - c2 * tsn[m];
      si += c2 * tc[m] + a * tsn[m];
    }
    Ar[i][r][f] = sr; Aq[i][r][f] = si;
  }
  __syncthreads();
  // irfft along columns (real part, hermitian weights), ortho 1/14; write u, s
  for (int e = tid; e < kINT * kNP; e += 128) {
    int i = e / kNP, ppp = e % kNP;
    int r = ppp / kH, c = ppp % kH;
    float acc = 0;
    for (int f = 0; f < kHF; ++f) {
      int m = (f * c) % kH;
      float w = (f == 0 || f == 7) ? 1.0f : 2.0f;
      acc += w * (Ar[i][r][f] * tc[m] - Aq[i][r][f] * tsn[m]);
    }
    acc *= (1.0f / 14.0f);
    size_t oi = (((size_t)b * kDIM + g) * kINT + i) * kNP + ppp;
    u[oi] = acc;
    s[oi] = softf(acc, kLAM);
  }
}

// ---------------- ISTA step: s = soft(s - (conv(s,kk) - u)/L, LAM/L) ----------------
__global__ __launch_bounds__(128) void conv_ista(const float* __restrict__ u,
                                                 float* __restrict__ s,
                                                 const float* __restrict__ kk,
                                                 const float* __restrict__ Lb)
{
  int b = blockIdx.x / kDIM, g = blockIdx.x % kDIM;
  __shared__ float tc[kH], tsn[kH];
  __shared__ float xs[kINT][kH][kH];
  __shared__ float Tr[kINT][kH][kHF], Tq[kINT][kH][kHF];
  __shared__ float Xr[kINT][kH][kHF], Xq[kINT][kH][kHF];
  __shared__ float Yr[kINT][kH][kHF], Yq[kINT][kH][kHF];
  int tid = threadIdx.x;
  build_tw(tc, tsn);
  const size_t base = ((size_t)b * kDIM + g) * kINT * kNP;
  for (int e = tid; e < kINT * kNP; e += 128) {
    int i = e / kNP, ppp = e % kNP;
    xs[i][ppp / kH][ppp % kH] = s[base + e];
  }
  __syncthreads();
  // rfft cols (5 channels)
  for (int e = tid; e < kINT * 112; e += 128) {
    int i = e / 112, rf = e % 112;
    int r = rf >> 3, f = rf & 7;
    float sr = 0, si = 0;
    for (int c = 0; c < kH; ++c) {
      int m = (f * c) % kH;
      float xv = xs[i][r][c];
      sr += xv * tc[m]; si -= xv * tsn[m];
    }
    Tr[i][r][f] = sr; Tq[i][r][f] = si;
  }
  __syncthreads();
  // fft rows
  for (int e = tid; e < kINT * 112; e += 128) {
    int i = e / 112, hf = e % 112;
    int h = hf >> 3, f = hf & 7;
    float sr = 0, si = 0;
    for (int r = 0; r < kH; ++r) {
      int m = (h * r) % kH;
      float a = Tr[i][r][f], c2 = Tq[i][r][f];
      sr += a * tc[m] + c2 * tsn[m];
      si += c2 * tc[m] - a * tsn[m];
    }
    Xr[i][h][f] = sr * (1.0f / 14.0f);
    Xq[i][h][f] = si * (1.0f / 14.0f);
  }
  __syncthreads();
  // Gram multiply: Y_i = sum_j kk[g,i,j] * X_j
  for (int e = tid; e < kINT * 112; e += 128) {
    int i = e / 112, hf = e % 112;
    int h = hf >> 3, f = hf & 7;
    const float* kbp = kk + ((size_t)g * 25 + i * 5) * 224 + (size_t)hf * 2;
    float sr = 0, si = 0;
#pragma unroll
    for (int j = 0; j < kINT; ++j) {
      float kr = kbp[j * 224], kq = kbp[j * 224 + 1];
      float xr = Xr[j][h][f], xq = Xq[j][h][f];
      sr += kr * xr - kq * xq;
      si += kr * xq + kq * xr;
    }
    Yr[i][h][f] = sr; Yq[i][h][f] = si;
  }
  __syncthreads();
  // ifft rows (reuse T arrays)
  for (int e = tid; e < kINT * 112; e += 128) {
    int i = e / 112, rf = e % 112;
    int r = rf >> 3, f = rf & 7;
    float sr = 0, si = 0;
    for (int h = 0; h < kH; ++h) {
      int m = (h * r) % kH;
      float a = Yr[i][h][f], c2 = Yq[i][h][f];
      sr += a * tc[m] - c2 * tsn[m];
      si += c2 * tc[m] + a * tsn[m];
    }
    Tr[i][r][f] = sr; Tq[i][r][f] = si;
  }
  __syncthreads();
  // irfft cols + ISTA update
  float Lg = Lb[g];
  float invL = 1.0f / Lg;
  float thr = kLAM * invL;
  for (int e = tid; e < kINT * kNP; e += 128) {
    int i = e / kNP, ppp = e % kNP;
    int r = ppp / kH, c = ppp % kH;
    float acc = 0;
    for (int f = 0; f < kHF; ++f) {
      int m = (f * c) % kH;
      float w = (f == 0 || f == 7) ? 1.0f : 2.0f;
      acc += w * (Tr[i][r][f] * tc[m] - Tq[i][r][f] * tsn[m]);
    }
    acc *= (1.0f / 14.0f);
    float sv = xs[i][r][c];
    float uv = u[base + e];
    float t = sv - (acc - uv) * invL;
    s[base + e] = softf(t, thr);
  }
}

// ---------------- reconstruct: recon = circconv(circconv(s, conj(k)^T), k2) ----------------
__global__ __launch_bounds__(128) void conv_recon(const float* __restrict__ s,
                                                  const float* __restrict__ kern,
                                                  const float* __restrict__ kern2,
                                                  float* __restrict__ recon)
{
  int b = blockIdx.x / kDIM, g = blockIdx.x % kDIM;
  __shared__ float tc[kH], tsn[kH];
  __shared__ float xs[kINT][kH][kH];
  __shared__ float Tr[kINT][kH][kHF], Tq[kINT][kH][kHF];
  __shared__ float Xr[kINT][kH][kHF], Xq[kINT][kH][kHF];
  __shared__ float Zr[kH][kHF], Zq[kH][kHF];
  __shared__ float Br[kH][kHF], Bq[kH][kHF];
  int tid = threadIdx.x;
  build_tw(tc, tsn);
  const size_t base = ((size_t)b * kDIM + g) * kINT * kNP;
  for (int e = tid; e < kINT * kNP; e += 128) {
    int i = e / kNP, ppp = e % kNP;
    xs[i][ppp / kH][ppp % kH] = s[base + e];
  }
  __syncthreads();
  for (int e = tid; e < kINT * 112; e += 128) {
    int i = e / 112, rf = e % 112;
    int r = rf >> 3, f = rf & 7;
    float sr = 0, si = 0;
    for (int c = 0; c < kH; ++c) {
      int m = (f * c) % kH;
      float xv = xs[i][r][c];
      sr += xv * tc[m]; si -= xv * tsn[m];
    }
    Tr[i][r][f] = sr; Tq[i][r][f] = si;
  }
  __syncthreads();
  for (int e = tid; e < kINT * 112; e += 128) {
    int i = e / 112, hf = e % 112;
    int h = hf >> 3, f = hf & 7;
    float sr = 0, si = 0;
    for (int r = 0; r < kH; ++r) {
      int m = (h * r) % kH;
      float a = Tr[i][r][f], c2 = Tq[i][r][f];
      sr += a * tc[m] + c2 * tsn[m];
      si += c2 * tc[m] - a * tsn[m];
    }
    Xr[i][h][f] = sr * (1.0f / 14.0f);
    Xq[i][h][f] = si * (1.0f / 14.0f);
  }
  __syncthreads();
  // Z = sum_j conj(k_j) * X_j
  for (int e = tid; e < 112; e += 128) {
    int h = e >> 3, f = e & 7;
    float sr = 0, si = 0;
#pragma unroll
    for (int j = 0; j < kINT; ++j) {
      size_t ki = ((size_t)g * kINT + j) * 224 + (size_t)e * 2;
      float kr = kern[ki], kq = kern[ki + 1];
      float xr = Xr[j][h][f], xq = Xq[j][h][f];
      sr += kr * xr + kq * xq;    // Re(conj(k)*x)
      si += kr * xq - kq * xr;    // Im(conj(k)*x)
    }
    Zr[h][f] = sr; Zq[h][f] = si;
  }
  __syncthreads();
  // hermitian symmetrization of f=0 and f=7 columns (irfft->rfft round trip)
  float nzr = 0, nzi = 0; int sh = -1, sf = 0;
  if (tid < 28) {
    sh = tid % kH; sf = (tid / kH) ? 7 : 0;
    int h2 = (kH - sh) % kH;
    nzr = 0.5f * (Zr[sh][sf] + Zr[h2][sf]);
    nzi = 0.5f * (Zq[sh][sf] - Zq[h2][sf]);
  }
  __syncthreads();
  if (sh >= 0) { Zr[sh][sf] = nzr; Zq[sh][sf] = nzi; }
  __syncthreads();
  // multiply by kernel2 spectrum (in place, elementwise)
  for (int e = tid; e < 112; e += 128) {
    int h = e >> 3, f = e & 7;
    size_t ki = ((size_t)g * 112 + e) * 2;
    float kr = kern2[ki], kq = kern2[ki + 1];
    float zr = Zr[h][f], zq = Zq[h][f];
    Zr[h][f] = kr * zr - kq * zq;
    Zq[h][f] = kr * zq + kq * zr;
  }
  __syncthreads();
  // ifft rows
  for (int e = tid; e < 112; e += 128) {
    int r = e >> 3, f = e & 7;
    float sr = 0, si = 0;
    for (int h = 0; h < kH; ++h) {
      int m = (h * r) % kH;
      float a = Zr[h][f], c2 = Zq[h][f];
      sr += a * tc[m] - c2 * tsn[m];
      si += c2 * tc[m] + a * tsn[m];
    }
    Br[r][f] = sr; Bq[r][f] = si;
  }
  __syncthreads();
  // irfft cols -> recon (B, C, Np)
  float* og = recon + ((size_t)b * kDIM + g) * kNP;
  for (int e = tid; e < kNP; e += 128) {
    int r = e / kH, c = e % kH;
    float acc = 0;
    for (int f = 0; f < kHF; ++f) {
      int m = (f * c) % kH;
      float w = (f == 0 || f == 7) ? 1.0f : 2.0f;
      acc += w * (Br[r][f] * tc[m] - Bq[r][f] * tsn[m]);
    }
    og[e] = acc * (1.0f / 14.0f);
  }
}

// ---------------- BatchNorm (training stats) + bf16 cast (+transpose) ----------------
// layoutCN=1: xin is (B, C, Np); layoutCN=0: xin is (M, C). out always (M, C) bf16.
__global__ __launch_bounds__(256) void bn_apply(const float* __restrict__ xin,
                                                const float* __restrict__ gamma,
                                                const float* __restrict__ beta,
                                                __bf16* __restrict__ out,
                                                int layoutCN)
{
  int c = blockIdx.x;
  int tid = threadIdx.x;
  float s1 = 0, s2 = 0;
  for (int i = tid; i < kM; i += 256) {
    int b = i / kNP, n = i % kNP;
    float v = layoutCN ? xin[((size_t)b * kDIM + c) * kNP + n]
                       : xin[(size_t)i * kDIM + c];
    s1 += v; s2 += v * v;
  }
  __shared__ float r1[256], r2[256];
  r1[tid] = s1; r2[tid] = s2;
  __syncthreads();
  for (int o = 128; o > 0; o >>= 1) {
    if (tid < o) { r1[tid] += r1[tid + o]; r2[tid] += r2[tid + o]; }
    __syncthreads();
  }
  float m   = r1[0] * (1.0f / (float)kM);
  float var = r2[0] * (1.0f / (float)kM) - m * m;
  float sc  = rsqrtf(var + 1e-5f) * gamma[c];
  float bt  = beta[c];
  for (int i = tid; i < kM; i += 256) {
    int b = i / kNP, n = i % kNP;
    float v = layoutCN ? xin[((size_t)b * kDIM + c) * kNP + n]
                       : xin[(size_t)i * kDIM + c];
    out[(size_t)i * kDIM + c] = (__bf16)((v - m) * sc + bt);
  }
}

// ---------------- launch ----------------
extern "C" void kernel_launch(void* const* d_in, const int* in_sizes, int n_in,
                              void* d_out, int out_size, void* d_ws, size_t ws_size,
                              hipStream_t stream) {
  (void)in_sizes; (void)n_in; (void)out_size; (void)ws_size;
  const float* x      = (const float*)d_in[0];
  const float* v_w    = (const float*)d_in[1];
  const float* proj_w = (const float*)d_in[2];
  const float* proj_b = (const float*)d_in[3];
  const float* kern   = (const float*)d_in[4];
  const float* kern2  = (const float*)d_in[5];
  const float* n1g    = (const float*)d_in[6];
  const float* n1b    = (const float*)d_in[7];
  const float* n2g    = (const float*)d_in[8];
  const float* n2b    = (const float*)d_in[9];
  const float* fc1w   = (const float*)d_in[10];
  const float* fc1b   = (const float*)d_in[11];
  const float* fc2w   = (const float*)d_in[12];
  const float* fc2b   = (const float*)d_in[13];
  const float* Lbuf   = (const float*)d_in[14];

  char* p = (char*)d_ws;
  size_t off = 0;
  auto take = [&](size_t bytes) -> void* {
    void* q = p + off;
    off = (off + bytes + 255) & ~(size_t)255;
    return q;
  };

  __bf16* actbf = (__bf16*)take((size_t)kM * kDIM * 2);   // x_bf16 -> bn1 -> bn2 (serial reuse)
  __bf16* hbf   = (__bf16*)take((size_t)kM * kHID * 2);   // gelu(fc1) activations
  __bf16* wv    = (__bf16*)take((size_t)kDIM * kDIM * 2);
  __bf16* wp    = (__bf16*)take((size_t)kDIM * kDIM * 2);
  __bf16* w1    = (__bf16*)take((size_t)kHID * kDIM * 2);
  __bf16* w2    = (__bf16*)take((size_t)kDIM * kHID * 2);
  float* vxrec  = (float*)take((size_t)kM * kDIM * 4);            // vx (B,C,Np) -> recon (reuse)
  float* ubuf   = (float*)take((size_t)kB * kDIM * kINT * kNP * 4); // u -> 'new' (reuse)
  float* sbuf   = (float*)take((size_t)kB * kDIM * kINT * kNP * 4);
  float* kkbuf  = (float*)take((size_t)kDIM * 25 * 112 * 2 * 4);
  float* newbuf = ubuf;

  // 1) bf16 conversions
  cvt_bf16<<<(kM * kDIM + 255) / 256, 256, 0, stream>>>(x, actbf, kM * kDIM);
  cvt_bf16<<<(kDIM * kDIM + 255) / 256, 256, 0, stream>>>(v_w, wv, kDIM * kDIM);
  cvt_bf16<<<(kDIM * kDIM + 255) / 256, 256, 0, stream>>>(proj_w, wp, kDIM * kDIM);
  cvt_bf16<<<(kHID * kDIM + 255) / 256, 256, 0, stream>>>(fc1w, w1, kHID * kDIM);
  cvt_bf16<<<(kHID * kDIM + 255) / 256, 256, 0, stream>>>(fc2w, w2, kHID * kDIM);

  // 2) Gram spectra kk = k_i conj(k_j)
  kk_compute<<<kDIM, 256, 0, stream>>>(kern, kkbuf);

  // 3) vx = x @ v_w^T  (written transposed to (B, C, Np))
  dim3 gNdim(kDIM / GBN, kM / GBM);
  gemm_wmma<<<gNdim, 256, 0, stream>>>(actbf, wv, nullptr, nullptr, vxrec, nullptr,
                                       kM, kDIM, kDIM, 0);

  // 4) u = conv(vx, k); s = soft(u, LAM)
  conv_fwd<<<kB * kDIM, 128, 0, stream>>>(vxrec, kern, ubuf, sbuf);

  // 5) 3 ISTA iterations (in-place on s)
  for (int it = 0; it < 3; ++it)
    conv_ista<<<kB * kDIM, 128, 0, stream>>>(ubuf, sbuf, kkbuf, Lbuf);

  // 6) reconstruct (fused conj(k)^T conv + kernel2 conv)
  conv_recon<<<kB * kDIM, 128, 0, stream>>>(sbuf, kern, kern2, vxrec);

  // 7) BN1 + transpose -> bf16
  bn_apply<<<kDIM, 256, 0, stream>>>(vxrec, n1g, n1b, actbf, 1);

  // 8) new = x + BN1 @ proj_w^T + proj_b
  gemm_wmma<<<gNdim, 256, 0, stream>>>(actbf, wp, proj_b, x, newbuf, nullptr,
                                       kM, kDIM, kDIM, 1);

  // 9) BN2 -> bf16
  bn_apply<<<kDIM, 256, 0, stream>>>(newbuf, n2g, n2b, actbf, 0);

  // 10) h = gelu(BN2 @ fc1^T + b1) -> bf16
  dim3 gHdim(kHID / GBN, kM / GBM);
  gemm_wmma<<<gHdim, 256, 0, stream>>>(actbf, w1, fc1b, nullptr, nullptr, hbf,
                                       kM, kHID, kDIM, 2);

  // 11) out = new + h @ fc2^T + b2
  gemm_wmma<<<gNdim, 256, 0, stream>>>(hbf, w2, fc2b, newbuf, (float*)d_out, nullptr,
                                       kM, kDIM, kHID, 1);
}
